// Cluster_kmeans_pp_23519240913025
// MI455X (gfx1250) — matrix operand order repaired and checked by reading, not compiled, not run on hardware
//
#include <hip/hip_runtime.h>

// Problem dims (B=4096, n=1024, C=64, L=64 -> D = C*L = 4096)
#define B_ROWS 4096
#define N_CODES 1024
#define D_FEAT 4096

#define EMA_OLD 0.01f
#define EMA_NEW 0.99f

typedef __bf16 v16bf __attribute__((ext_vector_type(16)));
typedef float  v8f   __attribute__((ext_vector_type(8)));

struct __align__(16) U4   { unsigned x, y, z, w; };
struct __align__(32) U4x2 { U4 lo, hi; };

static __device__ __forceinline__ unsigned short f32_to_bf16_rne(float f) {
  unsigned u = __float_as_uint(f);
  return (unsigned short)((u + 0x7FFFu + ((u >> 16) & 1u)) >> 16);
}
static __device__ __forceinline__ float bf16_bits_to_f32(unsigned short h) {
  return __uint_as_float(((unsigned)h) << 16);
}

// A fragment (16x32 bf16, ISA layout): lane<16 -> M=lane, K 0..7 & 16..23;
// lane>=16 -> M=lane-16, K 8..15 & 24..31. Two 16B LDS reads per fragment.
static __device__ __forceinline__ v16bf load_a_frag(const unsigned short* rowp, int koff0) {
  U4x2 t;
  t.lo = *(const U4*)(rowp + koff0);
  t.hi = *(const U4*)(rowp + 16 + koff0);
  return __builtin_bit_cast(v16bf, t);
}
// B fragment (32x16 bf16): lane<16 -> column N=lane holds K 0..7 & 16..23;
// lane>=16 -> column N=lane-16 holds K 8..15 & 24..31 (m is row-major [n][k]).
static __device__ __forceinline__ v16bf load_b_frag(const unsigned short* p) {
  U4x2 t;
  t.lo = *(const U4*)(p);
  t.hi = *(const U4*)(p + 16);
  return __builtin_bit_cast(v16bf, t);
}

__global__ void init_ws_kernel(unsigned long long* best, unsigned* winner) {
  int i = blockIdx.x * blockDim.x + threadIdx.x;
  if (i < B_ROWS) best[i] = ~0ull;
  if (i < N_CODES) winner[i] = 0u;
}

__global__ void mnorm_kernel(const float* __restrict__ m, float* __restrict__ mnorm) {
  __shared__ float red[256];
  int n = blockIdx.x;
  const float* row = m + (size_t)n * D_FEAT;
  float s = 0.f;
  for (int k = threadIdx.x * 4; k < D_FEAT; k += 256 * 4) {
    float4 v = *(const float4*)(row + k);
    s += v.x * v.x + v.y * v.y + v.z * v.z + v.w * v.w;
  }
  red[threadIdx.x] = s;
  __syncthreads();
  for (int off = 128; off > 0; off >>= 1) {
    if (threadIdx.x < off) red[threadIdx.x] += red[threadIdx.x + off];
    __syncthreads();
  }
  if (threadIdx.x == 0) mnorm[n] = red[0];
}

// One-time f32 -> (bf16_hi, bf16_lo) split into workspace (L2-resident).
__global__ void prepack_kernel(const float* __restrict__ src,
                               unsigned short* __restrict__ hi,
                               unsigned short* __restrict__ lo, int n4) {
  int i = blockIdx.x * blockDim.x + threadIdx.x;
  if (i >= n4) return;
  float4 v = ((const float4*)src)[i];
  unsigned short h0 = f32_to_bf16_rne(v.x), h1 = f32_to_bf16_rne(v.y);
  unsigned short h2 = f32_to_bf16_rne(v.z), h3 = f32_to_bf16_rne(v.w);
  unsigned short l0 = f32_to_bf16_rne(v.x - bf16_bits_to_f32(h0));
  unsigned short l1 = f32_to_bf16_rne(v.y - bf16_bits_to_f32(h1));
  unsigned short l2 = f32_to_bf16_rne(v.z - bf16_bits_to_f32(h2));
  unsigned short l3 = f32_to_bf16_rne(v.w - bf16_bits_to_f32(h3));
  uint2 hp, lp;
  hp.x = (unsigned)h0 | ((unsigned)h1 << 16);
  hp.y = (unsigned)h2 | ((unsigned)h3 << 16);
  lp.x = (unsigned)l0 | ((unsigned)l1 << 16);
  lp.y = (unsigned)l2 | ((unsigned)l3 << 16);
  *(uint2*)(hi + (size_t)i * 4) = hp;
  *(uint2*)(lo + (size_t)i * 4) = lp;
}

// ---------------- Fast path: prepacked bf16 + async-to-LDS + double buffer ----
#define BM 128
#define BN 128
#define KC 64
#define KPAD 72                     // padded K stride (elements)
#define ARR_ELEMS (BM * KPAD)       // 9216 shorts per array
#define BUF_ELEMS (4 * ARR_ELEMS)   // A_hi | A_lo | B_hi | B_lo

static __device__ __forceinline__ void async_b128(unsigned ldsoff, unsigned goff,
                                                  unsigned long long gbase) {
  asm volatile("global_load_async_to_lds_b128 %0, %1, %2"
               :: "v"(ldsoff), "v"(goff), "s"(gbase) : "memory");
}

__global__ __launch_bounds__(256)
void dist_argmin_fast_kernel(const unsigned short* __restrict__ y_hi,
                             const unsigned short* __restrict__ y_lo,
                             const unsigned short* __restrict__ m_hi,
                             const unsigned short* __restrict__ m_lo,
                             const float* __restrict__ mnorm,
                             unsigned long long* __restrict__ best) {
  __shared__ __align__(16) unsigned short lds[2 * BUF_ELEMS];  // 144 KB

  const int tid  = threadIdx.x;
  const int lane = tid & 31;
  const int wid  = tid >> 5;
  const int wm   = wid & 3;
  const int wn   = wid >> 2;
  const int b0   = blockIdx.x * BM;
  const int n0   = blockIdx.y * BN;

  const unsigned lbase = (unsigned)(unsigned long long)(void*)lds;  // LDS byte offset

  v8f acc[2][4];
#pragma unroll
  for (int i = 0; i < 2; ++i)
#pragma unroll
    for (int j = 0; j < 4; ++j) acc[i][j] = (v8f)(0.0f);

  unsigned long long srcs[4] = {(unsigned long long)y_hi, (unsigned long long)y_lo,
                                (unsigned long long)m_hi, (unsigned long long)m_lo};

  // Issue async copies for one KC-wide tile into buffer `buf`.
  // Per array: 128 rows x 128 B/row = 1024 x 16B chunks; 4 per thread.
  auto issue_tile = [&](int buf, int kc) {
#pragma unroll
    for (int arr = 0; arr < 4; ++arr) {
      int row0 = (arr < 2) ? b0 : n0;
#pragma unroll
      for (int q = 0; q < 4; ++q) {
        int chunk = tid + q * 256;
        int row = chunk >> 3, c = chunk & 7;
        unsigned goff = ((unsigned)(row0 + row) * D_FEAT + (unsigned)kc) * 2u +
                        (unsigned)c * 16u;
        unsigned loff = lbase + (unsigned)(buf * BUF_ELEMS + arr * ARR_ELEMS) * 2u +
                        (unsigned)(row * KPAD) * 2u + (unsigned)c * 16u;
        async_b128(loff, goff, srcs[arr]);
      }
    }
  };

  issue_tile(0, 0);
  const int NITER = D_FEAT / KC;
  for (int it = 0; it < NITER; ++it) {
    int cur = it & 1;
    bool more = (it + 1) < NITER;
    if (more) {
      issue_tile(cur ^ 1, (it + 1) * KC);
      asm volatile("s_wait_asynccnt 0x10" ::: "memory");  // older 16 copies done
    } else {
      asm volatile("s_wait_asynccnt 0x0" ::: "memory");
    }
    __syncthreads();

    const unsigned short* Ahi = lds + cur * BUF_ELEMS;
    const unsigned short* Alo = Ahi + ARR_ELEMS;
    const unsigned short* Bhi = Alo + ARR_ELEMS;
    const unsigned short* Blo = Bhi + ARR_ELEMS;

    const int arow   = wm * 32 + (lane & 15);
    const int koffA0 = (lane < 16) ? 0 : 8;
    const int brow   = wn * 64 + (lane & 15);
    const int koffB  = (lane < 16) ? 0 : 8;

#pragma unroll
    for (int ks = 0; ks < 2; ++ks) {        // two 32-wide sub-chunks
      const int kb = ks * 32;
      v16bf a_hi[2], a_lo[2];
#pragma unroll
      for (int i = 0; i < 2; ++i) {
        a_hi[i] = load_a_frag(Ahi + (arow + i * 16) * KPAD + kb, koffA0);
        a_lo[i] = load_a_frag(Alo + (arow + i * 16) * KPAD + kb, koffA0);
      }
      v16bf b_hi[4], b_lo[4];
#pragma unroll
      for (int j = 0; j < 4; ++j) {
        b_hi[j] = load_b_frag(Bhi + (brow + j * 16) * KPAD + kb + koffB);
        b_lo[j] = load_b_frag(Blo + (brow + j * 16) * KPAD + kb + koffB);
      }
#pragma unroll
      for (int i = 0; i < 2; ++i)
#pragma unroll
        for (int j = 0; j < 4; ++j) {
          acc[i][j] = __builtin_amdgcn_wmma_f32_16x16x32_bf16(
              false, a_hi[i], false, b_hi[j], (short)0, acc[i][j], false, false);
          acc[i][j] = __builtin_amdgcn_wmma_f32_16x16x32_bf16(
              false, a_hi[i], false, b_lo[j], (short)0, acc[i][j], false, false);
          acc[i][j] = __builtin_amdgcn_wmma_f32_16x16x32_bf16(
              false, a_lo[i], false, b_hi[j], (short)0, acc[i][j], false, false);
        }
    }
    __syncthreads();
  }

  // Epilogue: min over this wave's 64 columns, then global atomicMin.
#pragma unroll
  for (int i = 0; i < 2; ++i) {
#pragma unroll
    for (int r = 0; r < 8; ++r) {
      int Mloc = (lane < 16) ? r : (r + 8);
      int b = b0 + wm * 32 + i * 16 + Mloc;
      unsigned long long candbest = ~0ull;
#pragma unroll
      for (int j = 0; j < 4; ++j) {
        int n = n0 + wn * 64 + j * 16 + (lane & 15);
        float score = mnorm[n] - 2.0f * acc[i][j][r];
        unsigned s = __float_as_uint(score);
        unsigned key = (s & 0x80000000u) ? ~s : (s | 0x80000000u);
        unsigned long long cand = ((unsigned long long)key << 32) | (unsigned)n;
        candbest = cand < candbest ? cand : candbest;
      }
#pragma unroll
      for (int off = 1; off < 16; off <<= 1) {
        unsigned long long other = __shfl_xor(candbest, off, 16);
        candbest = other < candbest ? other : candbest;
      }
      if ((lane & 15) == 0) atomicMin(best + b, candbest);
    }
  }
}

// ---------------- Fallback path (small ws): convert-in-kernel, KC=32 ----------
#define KC1 32
#define KPAD1 40

static __device__ __forceinline__ void store_hi_lo(unsigned short* hiArr,
                                                   unsigned short* loArr,
                                                   int idx, float4 v) {
  unsigned short h0 = f32_to_bf16_rne(v.x), h1 = f32_to_bf16_rne(v.y);
  unsigned short h2 = f32_to_bf16_rne(v.z), h3 = f32_to_bf16_rne(v.w);
  unsigned short l0 = f32_to_bf16_rne(v.x - bf16_bits_to_f32(h0));
  unsigned short l1 = f32_to_bf16_rne(v.y - bf16_bits_to_f32(h1));
  unsigned short l2 = f32_to_bf16_rne(v.z - bf16_bits_to_f32(h2));
  unsigned short l3 = f32_to_bf16_rne(v.w - bf16_bits_to_f32(h3));
  uint2 hp, lp;
  hp.x = (unsigned)h0 | ((unsigned)h1 << 16);
  hp.y = (unsigned)h2 | ((unsigned)h3 << 16);
  lp.x = (unsigned)l0 | ((unsigned)l1 << 16);
  lp.y = (unsigned)l2 | ((unsigned)l3 << 16);
  *(uint2*)(hiArr + idx) = hp;
  *(uint2*)(loArr + idx) = lp;
}

__global__ __launch_bounds__(256)
void dist_argmin_fallback_kernel(const float* __restrict__ y, const float* __restrict__ m,
                                 const float* __restrict__ mnorm,
                                 unsigned long long* __restrict__ best) {
  __shared__ unsigned short As_hi[BM * KPAD1];
  __shared__ unsigned short As_lo[BM * KPAD1];
  __shared__ unsigned short Bs_hi[BN * KPAD1];
  __shared__ unsigned short Bs_lo[BN * KPAD1];

  const int tid  = threadIdx.x;
  const int lane = tid & 31;
  const int wid  = tid >> 5;
  const int wm   = wid & 3;
  const int wn   = wid >> 2;
  const int b0   = blockIdx.x * BM;
  const int n0   = blockIdx.y * BN;

  v8f acc[2][4];
#pragma unroll
  for (int i = 0; i < 2; ++i)
#pragma unroll
    for (int j = 0; j < 4; ++j) acc[i][j] = (v8f)(0.0f);

  const int rbase = tid >> 3;
  const int c4    = (tid & 7) * 4;

  for (int kc = 0; kc < D_FEAT; kc += KC1) {
#pragma unroll
    for (int it = 0; it < 4; ++it) {
      int r = rbase + it * 32;
      float4 v = *(const float4*)(y + (size_t)(b0 + r) * D_FEAT + kc + c4);
      store_hi_lo(As_hi, As_lo, r * KPAD1 + c4, v);
      float4 w = *(const float4*)(m + (size_t)(n0 + r) * D_FEAT + kc + c4);
      store_hi_lo(Bs_hi, Bs_lo, r * KPAD1 + c4, w);
    }
    __syncthreads();

    const int arow   = wm * 32 + (lane & 15);
    const int koffA0 = (lane < 16) ? 0 : 8;
    v16bf a_hi[2], a_lo[2];
#pragma unroll
    for (int i = 0; i < 2; ++i) {
      a_hi[i] = load_a_frag(As_hi + (arow + i * 16) * KPAD1, koffA0);
      a_lo[i] = load_a_frag(As_lo + (arow + i * 16) * KPAD1, koffA0);
    }
    const int brow  = wn * 64 + (lane & 15);
    const int koffB = (lane < 16) ? 0 : 8;
    v16bf b_hi[4], b_lo[4];
#pragma unroll
    for (int j = 0; j < 4; ++j) {
      b_hi[j] = load_b_frag(Bs_hi + (brow + j * 16) * KPAD1 + koffB);
      b_lo[j] = load_b_frag(Bs_lo + (brow + j * 16) * KPAD1 + koffB);
    }
#pragma unroll
    for (int i = 0; i < 2; ++i)
#pragma unroll
      for (int j = 0; j < 4; ++j) {
        acc[i][j] = __builtin_amdgcn_wmma_f32_16x16x32_bf16(
            false, a_hi[i], false, b_hi[j], (short)0, acc[i][j], false, false);
        acc[i][j] = __builtin_amdgcn_wmma_f32_16x16x32_bf16(
            false, a_hi[i], false, b_lo[j], (short)0, acc[i][j], false, false);
        acc[i][j] = __builtin_amdgcn_wmma_f32_16x16x32_bf16(
            false, a_lo[i], false, b_hi[j], (short)0, acc[i][j], false, false);
      }
    __syncthreads();
  }

#pragma unroll
  for (int i = 0; i < 2; ++i) {
#pragma unroll
    for (int r = 0; r < 8; ++r) {
      int Mloc = (lane < 16) ? r : (r + 8);
      int b = b0 + wm * 32 + i * 16 + Mloc;
      unsigned long long candbest = ~0ull;
#pragma unroll
      for (int j = 0; j < 4; ++j) {
        int n = n0 + wn * 64 + j * 16 + (lane & 15);
        float score = mnorm[n] - 2.0f * acc[i][j][r];
        unsigned s = __float_as_uint(score);
        unsigned key = (s & 0x80000000u) ? ~s : (s | 0x80000000u);
        unsigned long long cand = ((unsigned long long)key << 32) | (unsigned)n;
        candbest = cand < candbest ? cand : candbest;
      }
#pragma unroll
      for (int off = 1; off < 16; off <<= 1) {
        unsigned long long other = __shfl_xor(candbest, off, 16);
        candbest = other < candbest ? other : candbest;
      }
      if ((lane & 15) == 0) atomicMin(best + b, candbest);
    }
  }
}

// winner[c] = 1 + max{ b : z[b]==c } (0 = no writer) -> last-write-wins.
__global__ void assign_kernel(const unsigned long long* __restrict__ best,
                              unsigned* __restrict__ winner) {
  int b = blockIdx.x * blockDim.x + threadIdx.x;
  if (b >= B_ROWS) return;
  unsigned n = (unsigned)(best[b] & 0xFFFFFFFFull);
  atomicMax(winner + n, (unsigned)(b + 1));
}

__global__ void update_kernel(const float* __restrict__ y, const float* __restrict__ m,
                              const float* __restrict__ sd,
                              const unsigned* __restrict__ winner,
                              float* __restrict__ out) {
  const int D4 = D_FEAT / 4;
  size_t idx = (size_t)blockIdx.x * blockDim.x + threadIdx.x;
  if (idx >= (size_t)N_CODES * D4) return;
  int c = (int)(idx / D4);
  int i = (int)(idx % D4) * 4;
  float4 mv = *(const float4*)(m + (size_t)c * D_FEAT + i);
  float4 sv = *(const float4*)(sd + (size_t)c * D_FEAT + i);
  float4 om = mv, os = sv;
  unsigned w = winner[c];
  if (w) {
    int b = (int)w - 1;
    float4 yv = *(const float4*)(y + (size_t)b * D_FEAT + i);
    om.x = mv.x * EMA_OLD + yv.x * EMA_NEW;
    om.y = mv.y * EMA_OLD + yv.y * EMA_NEW;
    om.z = mv.z * EMA_OLD + yv.z * EMA_NEW;
    om.w = mv.w * EMA_OLD + yv.w * EMA_NEW;
    float dx = om.x - yv.x, dy = om.y - yv.y, dz = om.z - yv.z, dw = om.w - yv.w;
    os.x = dx * dx * EMA_OLD + sv.x * EMA_NEW;
    os.y = dy * dy * EMA_OLD + sv.y * EMA_NEW;
    os.z = dz * dz * EMA_OLD + sv.z * EMA_NEW;
    os.w = dw * dw * EMA_OLD + sv.w * EMA_NEW;
  }
  *(float4*)(out + (size_t)c * D_FEAT + i) = om;
  *(float4*)(out + (size_t)(N_CODES + c) * D_FEAT + i) = os;
}

extern "C" void kernel_launch(void* const* d_in, const int* in_sizes, int n_in,
                              void* d_out, int out_size, void* d_ws, size_t ws_size,
                              hipStream_t stream) {
  const float* y  = (const float*)d_in[0];
  const float* m  = (const float*)d_in[1];
  const float* sd = (const float*)d_in[2];
  // d_in[3] = p: unused by the reference output.

  char* ws = (char*)d_ws;
  unsigned long long* best = (unsigned long long*)ws;                      // 32 KB
  unsigned* winner = (unsigned*)(ws + (size_t)B_ROWS * 8);                 // 4 KB
  float* mnorm = (float*)(ws + (size_t)B_ROWS * 8 + (size_t)N_CODES * 4);  // 4 KB

  const size_t packOff = 65536;
  const size_t yBytes = (size_t)B_ROWS * D_FEAT * 2;    // 32 MB per array
  const size_t mBytes = (size_t)N_CODES * D_FEAT * 2;   // 8 MB per array
  const size_t need = packOff + 2 * yBytes + 2 * mBytes;

  init_ws_kernel<<<B_ROWS / 256, 256, 0, stream>>>(best, winner);
  mnorm_kernel<<<N_CODES, 256, 0, stream>>>(m, mnorm);

  dim3 grid(B_ROWS / BM, N_CODES / BN);
  if (ws_size >= need) {
    unsigned short* y_hi = (unsigned short*)(ws + packOff);
    unsigned short* y_lo = (unsigned short*)(ws + packOff + yBytes);
    unsigned short* m_hi = (unsigned short*)(ws + packOff + 2 * yBytes);
    unsigned short* m_lo = (unsigned short*)(ws + packOff + 2 * yBytes + mBytes);
    int y4 = B_ROWS * (D_FEAT / 4), m4 = N_CODES * (D_FEAT / 4);
    prepack_kernel<<<y4 / 256, 256, 0, stream>>>(y, y_hi, y_lo, y4);
    prepack_kernel<<<m4 / 256, 256, 0, stream>>>(m, m_hi, m_lo, m4);
    dist_argmin_fast_kernel<<<grid, 256, 0, stream>>>(y_hi, y_lo, m_hi, m_lo, mnorm, best);
  } else {
    dist_argmin_fallback_kernel<<<grid, 256, 0, stream>>>(y, m, mnorm, best);
  }

  assign_kernel<<<B_ROWS / 256, 256, 0, stream>>>(best, winner);
  update_kernel<<<(N_CODES * (D_FEAT / 4)) / 256, 256, 0, stream>>>(y, m, sd, winner,
                                                                    (float*)d_out);
}
